// LSHAttention_31619549233731
// MI455X (gfx1250) — matrix operand order repaired and checked by reading, not compiled, not run on hardware
//
#include <hip/hip_runtime.h>
#include <hip/hip_fp16.h>

typedef _Float16 v8h  __attribute__((ext_vector_type(8)));
typedef _Float16 v16h __attribute__((ext_vector_type(16)));
typedef float    v8f  __attribute__((ext_vector_type(8)));

constexpr int B   = 16;
constexpr int L   = 4096;
constexpr int DK  = 64;
constexpr int R   = 4;
constexpr int NBK = 64;   // buckets
constexpr int NB2 = 32;   // rand_matrix columns per round

// ---------- WMMA fragment helpers (CDNA5 16x16x32 f16, wave32) ----------
__device__ __forceinline__ v16h pack16(v8h lo, v8h hi) {
  v16h r;
#pragma unroll
  for (int i = 0; i < 8; ++i) { r[i] = lo[i]; r[i + 8] = hi[i]; }
  return r;
}
// B fragment (p = base + 16*hh): e<8 -> K = 8*hh+e ; e>=8 -> K = 16+8*hh+(e-8)
__device__ __forceinline__ v16h ld_b16(const _Float16* p) {
  const int bh = (threadIdx.x & 31) >> 4;
  v8h lo = *(const v8h*)(p - 8 * bh);
  v8h hi = *(const v8h*)(p + 16 - 8 * bh);
  return pack16(lo, hi);
}
// A fragment: e<8 -> K = kb+8*hh+e ; e>=8 -> K = kb+16+8*hh+(e-8)
__device__ __forceinline__ v16h ld_a16(const _Float16* rowp, int kb, int hh) {
  v8h lo = *(const v8h*)(rowp + kb + 8 * hh);
  v8h hi = *(const v8h*)(rowp + kb + 16 + 8 * hh);
  return pack16(lo, hi);
}
__device__ __forceinline__ v8f wmma16(v16h a, v16h b, v8f c) {
  return __builtin_amdgcn_wmma_f32_16x16x32_f16(false, a, false, b, (short)0, c,
                                                false, false);
}

// ---------- Kernel 1: LSH hashing (bucket ids) ----------
__global__ void __launch_bounds__(256) k_hash(const float* __restrict__ q,
                                              const float* __restrict__ rnd,
                                              int* __restrict__ bucket) {
  __shared__ float rmn[DK][NB2];
  __shared__ float cinv[NB2];
  const int b = blockIdx.x >> 2, r = blockIdx.x & 3;
  const int tid = threadIdx.x;
  for (int idx = tid; idx < DK * NB2; idx += 256) {
    int d = idx >> 5, nn = idx & 31;
    rmn[d][nn] = rnd[((b * DK + d) * R + r) * NB2 + nn];
  }
  __syncthreads();
  if (tid < NB2) {
    float s = 0.f;
    for (int d = 0; d < DK; ++d) { float x = rmn[d][tid]; s += x * x; }
    cinv[tid] = 1.0f / sqrtf(s);   // reference normalizes with no epsilon
  }
  __syncthreads();
  for (int idx = tid; idx < DK * NB2; idx += 256) {
    int d = idx >> 5, nn = idx & 31;
    rmn[d][nn] *= cinv[nn];
  }
  __syncthreads();
  for (int l = tid; l < L; l += 256) {
    float acc[NB2];
#pragma unroll
    for (int nn = 0; nn < NB2; ++nn) acc[nn] = 0.f;
    const float* qp = &q[((size_t)(b * L + l)) * DK];
    for (int d = 0; d < DK; ++d) {
      float qd = qp[d];
#pragma unroll
      for (int nn = 0; nn < NB2; ++nn) acc[nn] += qd * rmn[d][nn];
    }
    // argmax over concat([mm, -mm]); strict '>' => first-occurrence tie-break
    float best = acc[0]; int bi = 0;
#pragma unroll
    for (int nn = 1; nn < NB2; ++nn) if (acc[nn] > best) { best = acc[nn]; bi = nn; }
#pragma unroll
    for (int nn = 0; nn < NB2; ++nn) if (-acc[nn] > best) { best = -acc[nn]; bi = NB2 + nn; }
    bucket[(b * R + r) * L + l] = bi;
  }
}

// ---------- Kernel 2: stable counting sort per (b,r) ----------
__global__ void __launch_bounds__(256) k_sort(const int* __restrict__ bucket,
                                              int* __restrict__ sort_i,
                                              int* __restrict__ sbkt,
                                              int* __restrict__ inv_i) {
  __shared__ int buf[L];
  __shared__ int hist[NBK];
  __shared__ int start[NBK];
  const int base = blockIdx.x * L;
  const int tid = threadIdx.x;
  if (tid < NBK) hist[tid] = 0;
  __syncthreads();
  for (int l = tid; l < L; l += 256) {
    int bv = bucket[base + l];
    buf[l] = bv;
    atomicAdd(&hist[bv], 1);
  }
  __syncthreads();
  if (tid == 0) {
    int run = 0;
    for (int k = 0; k < NBK; ++k) { start[k] = run; run += hist[k]; }
  }
  __syncthreads();
  if (tid < NBK) {         // one thread per bucket: stable scatter by position
    int o = start[tid];
    for (int l = 0; l < L; ++l) {
      if (buf[l] == tid) {
        sort_i[base + o] = l;
        sbkt[base + o]   = tid;
        inv_i[base + l]  = o;
        ++o;
      }
    }
  }
}

// ---------- Kernel 3: per-(b,bucket-tile,round) attention with WMMA ----------
__global__ void __launch_bounds__(128) k_attn(const float* __restrict__ q,
                                              const float* __restrict__ val,
                                              const int* __restrict__ sort_i,
                                              const int* __restrict__ sbkt,
                                              const int* __restrict__ inv_i,
                                              _Float16* __restrict__ att_o,
                                              float* __restrict__ lse_o) {
  // LDS layout (45,824 B): Qh[64][64]h | Kh[128][64]h / Ph[64][128]h |
  //                        Vt[64][128]h / attF[64][64]f | meta | lse_row
  constexpr int QH_OFF = 0,     KHP_OFF = 8192,  VT_OFF = 24576;
  constexpr int QI_OFF = 40960, BQ_OFF = 41216,  KI_OFF = 41472;
  constexpr int BK_OFF = 41984, QN2_OFF = 42496, KB2_OFF = 43520, LSE_OFF = 45568;
  __shared__ __align__(32) unsigned char smem[45824];
  _Float16* Qh   = (_Float16*)(smem + QH_OFF);
  _Float16* Kh   = (_Float16*)(smem + KHP_OFF);
  _Float16* Ph   = (_Float16*)(smem + KHP_OFF);   // aliases Kh (dead after scores)
  _Float16* Vt   = (_Float16*)(smem + VT_OFF);
  float*    attF = (float*)(smem + VT_OFF);       // aliases Vt (dead after PV)
  int*   qi  = (int*)(smem + QI_OFF);
  int*   bq  = (int*)(smem + BQ_OFF);
  int*   ki  = (int*)(smem + KI_OFF);
  int*   bk  = (int*)(smem + BK_OFF);
  int*   qn2 = (int*)(smem + QN2_OFF);            // [64][4]
  int*   kb2 = (int*)(smem + KB2_OFF);            // [128][4]
  float* lse_row = (float*)(smem + LSE_OFF);

  const int nt = blockIdx.x, r = blockIdx.y, b = blockIdx.z;
  const int tid = threadIdx.x, lane = tid & 31, wv = tid >> 5;
  const int hh = lane >> 4, ln = lane & 15;
  const int base = (b * R + r) * L;

  // ---- metadata: rows (queries) and window keys ----
  if (tid < 64) {
    int rank = nt * 64 + tid;
    int l = sort_i[base + rank];
    qi[tid] = l; bq[tid] = sbkt[base + rank];
#pragma unroll
    for (int rp = 0; rp < R; ++rp) qn2[tid * 4 + rp] = inv_i[(b * R + rp) * L + l] >> 6;
  }
  {
    int j = tid;
    int krank = (j < 64) ? (((nt + NBK - 1) & (NBK - 1)) * 64 + j)
                         : (nt * 64 + j - 64);
    int l = sort_i[base + krank];
    ki[j] = l; bk[j] = sbkt[base + krank];
#pragma unroll
    for (int rp = 0; rp < R; ++rp) kb2[j * 4 + rp] = inv_i[(b * R + rp) * L + l] >> 6;
  }
  __syncthreads();

  // ---- gather Q (raw, f16) ----
  for (int idx = tid; idx < 64 * 16; idx += 128) {
    int row = idx >> 4, c = (idx & 15) * 4;
    float4 qv = *(const float4*)&q[((size_t)(b * L + qi[row])) * DK + c];
    Qh[row * 64 + c + 0] = (_Float16)qv.x;
    Qh[row * 64 + c + 1] = (_Float16)qv.y;
    Qh[row * 64 + c + 2] = (_Float16)qv.z;
    Qh[row * 64 + c + 3] = (_Float16)qv.w;
  }
  // ---- gather K (l2-normalized) and V (transposed) ----
  {
    int j = tid;
    const float* kp = &q[((size_t)(b * L + ki[j])) * DK];
    float ss = 0.f;
#pragma unroll
    for (int c = 0; c < 16; ++c) {
      float4 kv = *(const float4*)(kp + 4 * c);
      ss += kv.x * kv.x + kv.y * kv.y + kv.z * kv.z + kv.w * kv.w;
    }
    float rn = 1.0f / fmaxf(sqrtf(ss), 1e-12f);
#pragma unroll
    for (int c = 0; c < 16; ++c) {
      float4 kv = *(const float4*)(kp + 4 * c);
      Kh[j * 64 + 4 * c + 0] = (_Float16)(kv.x * rn);
      Kh[j * 64 + 4 * c + 1] = (_Float16)(kv.y * rn);
      Kh[j * 64 + 4 * c + 2] = (_Float16)(kv.z * rn);
      Kh[j * 64 + 4 * c + 3] = (_Float16)(kv.w * rn);
    }
    const float* vp = &val[((size_t)(b * L + ki[j])) * DK];
#pragma unroll
    for (int c = 0; c < 16; ++c) {
      float4 vv = *(const float4*)(vp + 4 * c);
      Vt[(4 * c + 0) * 128 + j] = (_Float16)vv.x;
      Vt[(4 * c + 1) * 128 + j] = (_Float16)vv.y;
      Vt[(4 * c + 2) * 128 + j] = (_Float16)vv.z;
      Vt[(4 * c + 3) * 128 + j] = (_Float16)vv.w;
    }
  }
  __syncthreads();

  // ---- scores: wave wv owns rows [16wv,16wv+16); S = Q . Khat^T ----
  v8f accS[8];
  {
    const _Float16* arow = Qh + (16 * wv + ln) * 64;
    v16h a0 = ld_a16(arow, 0, hh);
    v16h a1 = ld_a16(arow, 32, hh);
#pragma unroll
    for (int tl = 0; tl < 8; ++tl) {
      const _Float16* brow = Kh + (16 * tl + ln) * 64;
      v8f c = {};
      c = wmma16(a0, ld_b16(brow + 0 + 16 * hh), c);
      c = wmma16(a1, ld_b16(brow + 32 + 16 * hh), c);
      accS[tl] = c;
    }
  }
  __syncthreads();   // all Kh reads done (Ph will overwrite that region)

  // ---- mask + in-register softmax (row m = 16wv+8hh+v lives in a 16-lane half) ----
  int rqi[8], rbq[8], rqn[8][4];
#pragma unroll
  for (int vv = 0; vv < 8; ++vv) {
    int grow = 16 * wv + 8 * hh + vv;
    rqi[vv] = qi[grow]; rbq[vv] = bq[grow];
#pragma unroll
    for (int rp = 0; rp < 4; ++rp) rqn[vv][rp] = qn2[grow * 4 + rp];
  }
  float mx[8];
#pragma unroll
  for (int vv = 0; vv < 8; ++vv) mx[vv] = -3.0e38f;
#pragma unroll
  for (int tl = 0; tl < 8; ++tl) {
    int col = 16 * tl + ln;
    int cki = ki[col], cbk = bk[col];
#pragma unroll
    for (int vv = 0; vv < 8; ++vv) {
      float s = accS[tl][vv];
      if (rqi[vv] == cki)                        s = -1.0e5f;   // self
      else if (rbq[vv] != cbk || rqi[vv] < cki)  s = -1.0e9f;   // mismatch/causal
      else                                       s *= 0.125f;   // 1/sqrt(dk)
      accS[tl][vv] = s;
      mx[vv] = fmaxf(mx[vv], s);
    }
  }
#pragma unroll
  for (int vv = 0; vv < 8; ++vv) {
    float m = mx[vv];
    m = fmaxf(m, __shfl_xor(m, 1, 32));
    m = fmaxf(m, __shfl_xor(m, 2, 32));
    m = fmaxf(m, __shfl_xor(m, 4, 32));
    m = fmaxf(m, __shfl_xor(m, 8, 32));
    mx[vv] = m;
  }
  float sme[8];
#pragma unroll
  for (int vv = 0; vv < 8; ++vv) sme[vv] = 0.f;
#pragma unroll
  for (int tl = 0; tl < 8; ++tl)
#pragma unroll
    for (int vv = 0; vv < 8; ++vv) sme[vv] += expf(accS[tl][vv] - mx[vv]);
#pragma unroll
  for (int vv = 0; vv < 8; ++vv) {
    float s = sme[vv];
    s += __shfl_xor(s, 1, 32);
    s += __shfl_xor(s, 2, 32);
    s += __shfl_xor(s, 4, 32);
    s += __shfl_xor(s, 8, 32);
    sme[vv] = s;
  }
  if (ln == 0) {
#pragma unroll
    for (int vv = 0; vv < 8; ++vv)
      lse_row[16 * wv + 8 * hh + vv] = mx[vv] + logf(sme[vv]);
  }
  // p = exp(m - lse) / dup_count ; dup count = 1 + cross-round window membership
#pragma unroll
  for (int tl = 0; tl < 8; ++tl) {
    int col = 16 * tl + ln;
    int ckb[4];
#pragma unroll
    for (int rp = 0; rp < 4; ++rp) ckb[rp] = kb2[col * 4 + rp];
#pragma unroll
    for (int vv = 0; vv < 8; ++vv) {
      int cnt = 1;
      for (int rp = 0; rp < 4; ++rp) {
        if (rp == r) continue;
        int qb = rqn[vv][rp], kv = ckb[rp];
        cnt += (kv == qb || kv == ((qb + NBK - 1) & (NBK - 1))) ? 1 : 0;
      }
      float p = expf(accS[tl][vv] - mx[vv]) / (sme[vv] * (float)cnt);
      Ph[(16 * wv + 8 * hh + vv) * 128 + col] = (_Float16)p;
    }
  }
  __syncthreads();

  // ---- att = P . V ----
  v8f accO[4];
  {
    const _Float16* prow = Ph + (16 * wv + ln) * 128;
    v16h pa0 = ld_a16(prow, 0, hh);
    v16h pa1 = ld_a16(prow, 32, hh);
    v16h pa2 = ld_a16(prow, 64, hh);
    v16h pa3 = ld_a16(prow, 96, hh);
#pragma unroll
    for (int ntl = 0; ntl < 4; ++ntl) {
      const _Float16* vrow = Vt + (16 * ntl + ln) * 128;
      v8f c = {};
      c = wmma16(pa0, ld_b16(vrow + 0 + 16 * hh), c);
      c = wmma16(pa1, ld_b16(vrow + 32 + 16 * hh), c);
      c = wmma16(pa2, ld_b16(vrow + 64 + 16 * hh), c);
      c = wmma16(pa3, ld_b16(vrow + 96 + 16 * hh), c);
      accO[ntl] = c;
    }
  }
  __syncthreads();   // Vt reads done (attF will overwrite that region)
#pragma unroll
  for (int ntl = 0; ntl < 4; ++ntl)
#pragma unroll
    for (int vv = 0; vv < 8; ++vv)
      attF[(16 * wv + 8 * hh + vv) * 64 + 16 * ntl + ln] = accO[ntl][vv];
  __syncthreads();

  // ---- scatter to original order ----
  for (int idx = tid; idx < 64 * 64; idx += 128) {
    int row = idx >> 6, d = idx & 63;
    att_o[(((size_t)(b * L + qi[row])) * R + r) * DK + d] = (_Float16)attF[idx];
  }
  if (tid < 64) lse_o[base + qi[tid]] = lse_row[tid];
}

// ---------- Kernel 4: per-(b,r) softmax stats of lse over L ----------
__global__ void __launch_bounds__(256) k_lsered(const float* __restrict__ lse_o,
                                                float* __restrict__ red) {
  __shared__ float sb[256];
  const int br = blockIdx.x, tid = threadIdx.x;
  float mx = -3.0e38f;
  for (int k = tid; k < L; k += 256) mx = fmaxf(mx, lse_o[br * L + k]);
  sb[tid] = mx; __syncthreads();
  for (int s = 128; s > 0; s >>= 1) {
    if (tid < s) sb[tid] = fmaxf(sb[tid], sb[tid + s]);
    __syncthreads();
  }
  float gmax = sb[0]; __syncthreads();
  float sum = 0.f;
  for (int k = tid; k < L; k += 256) sum += expf(lse_o[br * L + k] - gmax);
  sb[tid] = sum; __syncthreads();
  for (int s = 128; s > 0; s >>= 1) {
    if (tid < s) sb[tid] += sb[tid + s];
    __syncthreads();
  }
  if (tid == 0) { red[br] = gmax; red[64 + br] = sb[0]; }
}

// ---------- Kernel 5: combine rounds with softmax(lse over L) weights ----------
__global__ void __launch_bounds__(256) k_out(const _Float16* __restrict__ att_o,
                                             const float* __restrict__ lse_o,
                                             const float* __restrict__ red,
                                             float* __restrict__ out) {
  int t = blockIdx.x * 256 + threadIdx.x;
  int d = t & 63;
  int l = (t >> 6) & (L - 1);
  int b = t >> 18;
  float acc = 0.f;
#pragma unroll
  for (int rr = 0; rr < R; ++rr) {
    int br = b * R + rr;
    float w = expf(lse_o[br * L + l] - red[br]) / red[64 + br];
    acc += (float)att_o[(((size_t)(b * L + l)) * R + rr) * DK + d] * w;
  }
  out[t] = acc;
}

extern "C" void kernel_launch(void* const* d_in, const int* in_sizes, int n_in,
                              void* d_out, int out_size, void* d_ws, size_t ws_size,
                              hipStream_t stream) {
  const float* q   = (const float*)d_in[0];
  const float* v   = (const float*)d_in[1];
  const float* rnd = (const float*)d_in[2];
  (void)in_sizes; (void)n_in; (void)out_size; (void)ws_size;  // seed unused (eval)

  char* ws = (char*)d_ws;
  int*      bucket = (int*)(ws + 0);                 // B*R*L            (1 MB)
  int*      sort_i = (int*)(ws + 1048576);           // sorted rank -> l (1 MB)
  int*      sbkt   = (int*)(ws + 2097152);           // bucket at rank   (1 MB)
  int*      inv_i  = (int*)(ws + 3145728);           // l -> rank        (1 MB)
  float*    lse_o  = (float*)(ws + 4194304);         // [b][r][l]        (1 MB)
  float*    red    = (float*)(ws + 5242880);         // max|sum per (b,r)
  _Float16* att_o  = (_Float16*)(ws + 5243392);      // [b][l][r][d]     (32 MB)
  float* out = (float*)d_out;

  k_hash <<<dim3(B * R),      dim3(256), 0, stream>>>(q, rnd, bucket);
  k_sort <<<dim3(B * R),      dim3(256), 0, stream>>>(bucket, sort_i, sbkt, inv_i);
  k_attn <<<dim3(NBK, R, B),  dim3(128), 0, stream>>>(q, v, sort_i, sbkt, inv_i,
                                                      att_o, lse_o);
  k_lsered<<<dim3(B * R),     dim3(256), 0, stream>>>(lse_o, red);
  k_out  <<<dim3(B * L * DK / 256), dim3(256), 0, stream>>>(att_o, lse_o, red, out);
}
